// FreqQuantize_11038065951109
// MI455X (gfx1250) — matrix operand order repaired and checked by reading, not compiled, not run on hardware
//
#include <hip/hip_runtime.h>
#include <math.h>
#include <stdint.h>

// ---------------------------------------------------------------------------
// FreqQuantize for MI455X (gfx1250, wave32, WMMA)
//   rows = B*C*N = 16*19*64 = 19456 ; IN=200 ; FREQ=101 ; D=64 ; CODES=8192
//   Dominant op: fused (19456x64)@(64x8192) GEMM + row argmax, x2 codebooks,
//   on v_wmma_f32_16x16x32_bf16. Codebook tiles are staged into LDS (shared
//   by the 4 waves of a block) with double-buffered async global->LDS copies.
// ---------------------------------------------------------------------------

typedef __bf16 bf16_t;
typedef __attribute__((ext_vector_type(16))) __bf16 v16bf;
typedef __attribute__((ext_vector_type(8)))  __bf16 v8bf;
typedef __attribute__((ext_vector_type(8)))  float  v8f;
typedef __attribute__((ext_vector_type(4)))  int    v4i;

#define ROWS   19456
#define IN_DIM 200
#define FREQ   101
#define VQD    64
#define NCODE  8192
#define RTILES (ROWS / 16)            // 1216 row tiles (exact)
#define WAVES_PER_BLOCK 4

#define CODES_PER_STAGE 64
#define TILES_PER_STAGE (CODES_PER_STAGE / 16)        // 4
#define NSTAGES (NCODE / CODES_PER_STAGE)             // 128
#define STAGE_ELEMS (CODES_PER_STAGE * VQD)           // 4096 bf16 per codebook
#define STAGE_BYTES (STAGE_ELEMS * 2)                 // 8192 B per codebook

#define AS1 __attribute__((address_space(1)))
#define AS3 __attribute__((address_space(3)))

#if __has_builtin(__builtin_amdgcn_global_load_async_to_lds_b128)
#define HAVE_ASYNC_LDS 1
#endif

#if __has_builtin(__builtin_amdgcn_s_wait_asynccnt)
#define WAIT_ASYNC(n) __builtin_amdgcn_s_wait_asynccnt(n)
#else
#define WAIT_ASYNC(n) asm volatile("s_wait_asynccnt %0" :: "i"(n) : "memory")
#endif

__device__ __forceinline__ unsigned short f2bf(float f) {
  union { float f; unsigned u; } v; v.f = f;
  unsigned u = v.u + 0x7FFFu + ((v.u >> 16) & 1u);   // round-to-nearest-even
  return (unsigned short)(u >> 16);
}

// --- Kernel 0: scaled DFT tables. cosT = cos(-2*pi*t*f/200)/sqrt(200),
//               sinT = sin(-2*pi*t*f/200)/sqrt(200)  (angle reduced mod 200).
__global__ void build_dft_tables(float* __restrict__ cosT, float* __restrict__ sinT) {
  int i = blockIdx.x * blockDim.x + threadIdx.x;
  if (i >= IN_DIM * FREQ) return;
  int t = i / FREQ;
  int f = i - t * FREQ;
  int m = (t * f) % IN_DIM;
  float ang = -6.28318530717958647692f * (float)m / (float)IN_DIM;
  float s, c;
  sincosf(ang, &s, &c);
  const float scale = 0.07071067811865475244f;   // 1/sqrt(200)
  cosT[i] = c * scale;
  sinT[i] = s * scale;
}

// --- Kernel 1: per-row DFT -> amp/phase -> 64-dim projections (bf16 out).
__global__ __launch_bounds__(128) void dft_project(
    const float* __restrict__ x,
    const float* __restrict__ projA, const float* __restrict__ projP,
    const float* __restrict__ cosT,  const float* __restrict__ sinT,
    unsigned short* __restrict__ featA16, unsigned short* __restrict__ featP16) {
  __shared__ float xs[IN_DIM];
  __shared__ float ampS[FREQ];
  __shared__ float phS[FREQ];
  const int row = blockIdx.x;
  const int tid = threadIdx.x;

  for (int i = tid; i < IN_DIM; i += 128) xs[i] = x[(size_t)row * IN_DIM + i];
  __syncthreads();

  if (tid < FREQ) {
    float R = 0.f, I = 0.f;
    #pragma unroll 4
    for (int t = 0; t < IN_DIM; ++t) {
      float xv = xs[t];
      R = fmaf(xv, cosT[t * FREQ + tid], R);
      I = fmaf(xv, sinT[t * FREQ + tid], I);
    }
    ampS[tid] = sqrtf(R * R + I * I);
    phS[tid]  = atan2f(I, R);
  }
  __syncthreads();

  if (tid < VQD) {
    float fa = 0.f, fp = 0.f;
    #pragma unroll 4
    for (int f = 0; f < FREQ; ++f) {
      fa = fmaf(ampS[f], projA[f * VQD + tid], fa);
      fp = fmaf(phS[f],  projP[f * VQD + tid], fp);
    }
    // L2-normalizing feat is argmax-invariant -> skipped on purpose.
    featA16[(size_t)row * VQD + tid] = f2bf(fa);
    featP16[(size_t)row * VQD + tid] = f2bf(fp);
  }
}

// --- Kernel 2: L2-normalize a codebook row, emit bf16.
__global__ void normalize_codebook(const float* __restrict__ cb,
                                   unsigned short* __restrict__ out, int M) {
  int m = blockIdx.x * blockDim.x + threadIdx.x;
  if (m >= M) return;
  const float4* p = (const float4*)(cb + (size_t)m * VQD);
  float v[VQD];
  float ss = 0.f;
  #pragma unroll
  for (int i = 0; i < VQD / 4; ++i) {
    float4 q = p[i];
    v[4*i+0] = q.x; v[4*i+1] = q.y; v[4*i+2] = q.z; v[4*i+3] = q.w;
    ss += q.x*q.x + q.y*q.y + q.z*q.z + q.w*q.w;
  }
  float inv = rsqrtf(ss);
  #pragma unroll
  for (int i = 0; i < VQD; ++i) out[(size_t)m * VQD + i] = f2bf(v[i] * inv);
}

// --- Cooperative stage copy: 16 KB (64 codes x 128 B x 2 codebooks) into LDS.
//     128 threads x 16 B x (4 + 4) chunks. Async path issues 8 b128s/thread
//     tracked on ASYNCcnt; fallback is plain global->VGPR->ds_store.
__device__ __forceinline__ void stage_issue(
    const unsigned short* __restrict__ cbA16, const unsigned short* __restrict__ cbP16,
    unsigned short* dstA, unsigned short* dstP, int stage, int tid) {
  const char* srcA = (const char*)cbA16 + (size_t)stage * STAGE_BYTES;
  const char* srcP = (const char*)cbP16 + (size_t)stage * STAGE_BYTES;
  char* dA = (char*)dstA;
  char* dP = (char*)dstP;
#ifdef HAVE_ASYNC_LDS
  #pragma unroll
  for (int j = 0; j < 4; ++j) {
    const int off = tid * 16 + j * 2048;
    __builtin_amdgcn_global_load_async_to_lds_b128(
        (AS1 v4i*)(uintptr_t)(srcA + off),
        (AS3 v4i*)(unsigned)(uintptr_t)(dA + off), 0, 0);
    __builtin_amdgcn_global_load_async_to_lds_b128(
        (AS1 v4i*)(uintptr_t)(srcP + off),
        (AS3 v4i*)(unsigned)(uintptr_t)(dP + off), 0, 0);
  }
#else
  #pragma unroll
  for (int j = 0; j < 4; ++j) {
    const int off = tid * 16 + j * 2048;
    *(uint4*)(dA + off) = *(const uint4*)(srcA + off);
    *(uint4*)(dP + off) = *(const uint4*)(srcP + off);
  }
#endif
}

// --- Kernel 3: fused bf16-WMMA GEMM + argmax, LDS-staged codebook tiles.
//  One wave per 16-row tile; 4 waves/block share double-buffered code stages.
__global__ __launch_bounds__(WAVES_PER_BLOCK * 32) void wmma_argmax(
    const unsigned short* __restrict__ featA16, const unsigned short* __restrict__ featP16,
    const unsigned short* __restrict__ cbA16,   const unsigned short* __restrict__ cbP16,
    float* __restrict__ outA, float* __restrict__ outP) {
  // [buffer][codebook][stage elems] = 2*2*4096 bf16 = 32 KB
  __shared__ unsigned short ldsbuf[2][2][STAGE_ELEMS];

  const int tid  = threadIdx.x;
  const int lane = tid & 31;
  const int wave = tid >> 5;
  const int row0 = (blockIdx.x * WAVES_PER_BLOCK + wave) * 16;
  const int h    = lane >> 4;    // half-wave id
  const int l15  = lane & 15;

  const bf16_t* fA = (const bf16_t*)featA16;
  const bf16_t* fP = (const bf16_t*)featP16;

  // A layout (16-bit A 16x32): lane -> row l15; element e -> K =
  //   kb*32 + (e>=8 ? 16 : 0) + 8*h + (e&7). Two contiguous 8-elem chunks.
  v16bf aA[2], aP[2];
  #pragma unroll
  for (int kb = 0; kb < 2; ++kb) {
    const size_t base = (size_t)(row0 + l15) * VQD + kb * 32 + 8 * h;
    v8bf a0 = *(const v8bf*)(fA + base);
    v8bf a1 = *(const v8bf*)(fA + base + 16);
    v8bf p0 = *(const v8bf*)(fP + base);
    v8bf p1 = *(const v8bf*)(fP + base + 16);
    #pragma unroll
    for (int e = 0; e < 8; ++e) {
      aA[kb][e] = a0[e]; aA[kb][e + 8] = a1[e];
      aP[kb][e] = p0[e]; aP[kb][e + 8] = p1[e];
    }
  }

  float bestA[8], bestP[8];
  int   idxA[8], idxP[8];
  #pragma unroll
  for (int r = 0; r < 8; ++r) {
    bestA[r] = -3.402823466e38f; bestP[r] = -3.402823466e38f;
    idxA[r] = 0; idxP[r] = 0;
  }

  // Prime stage 0.
  stage_issue(cbA16, cbP16, &ldsbuf[0][0][0], &ldsbuf[0][1][0], 0, tid);

  for (int s = 0; s < NSTAGES; ++s) {
    const int b = s & 1;
    if (s + 1 < NSTAGES) {
      stage_issue(cbA16, cbP16, &ldsbuf[b ^ 1][0][0], &ldsbuf[b ^ 1][1][0], s + 1, tid);
      WAIT_ASYNC(8);          // stage s complete; stage s+1 still in flight
    } else {
      WAIT_ASYNC(0);
    }
    __syncthreads();

    const bf16_t* lA = (const bf16_t*)&ldsbuf[b][0][0];
    const bf16_t* lP = (const bf16_t*)&ldsbuf[b][1][0];

    #pragma unroll
    for (int tt = 0; tt < TILES_PER_STAGE; ++tt) {
      const int n = s * CODES_PER_STAGE + tt * 16 + l15;   // this lane's code id
      // B layout (16-bit B 32x16): lane -> col l15; element e -> K = kb*32 + (e>=8 ? 16 : 0) + 8*h + (e&7).
      const int bb = (tt * 16 + l15) * VQD + 8 * h;
      v16bf bA0, bA1, bP0, bP1;
      {
        v8bf x0 = *(const v8bf*)(lA + bb), x1 = *(const v8bf*)(lA + bb + 16);
        v8bf x2 = *(const v8bf*)(lA + bb + 32), x3 = *(const v8bf*)(lA + bb + 48);
        v8bf y0 = *(const v8bf*)(lP + bb), y1 = *(const v8bf*)(lP + bb + 16);
        v8bf y2 = *(const v8bf*)(lP + bb + 32), y3 = *(const v8bf*)(lP + bb + 48);
        #pragma unroll
        for (int e = 0; e < 8; ++e) {
          bA0[e] = x0[e]; bA0[e + 8] = x1[e]; bA1[e] = x2[e]; bA1[e + 8] = x3[e];
          bP0[e] = y0[e]; bP0[e + 8] = y1[e]; bP1[e] = y2[e]; bP1[e + 8] = y3[e];
        }
      }

      v8f sA = {};
      sA = __builtin_amdgcn_wmma_f32_16x16x32_bf16(false, aA[0], false, bA0, (short)0, sA, false, false);
      sA = __builtin_amdgcn_wmma_f32_16x16x32_bf16(false, aA[1], false, bA1, (short)0, sA, false, false);
      v8f sP = {};
      sP = __builtin_amdgcn_wmma_f32_16x16x32_bf16(false, aP[0], false, bP0, (short)0, sP, false, false);
      sP = __builtin_amdgcn_wmma_f32_16x16x32_bf16(false, aP[1], false, bP1, (short)0, sP, false, false);

      // C layout: VGPR r -> row (r + 8*h), column l15.
      #pragma unroll
      for (int r = 0; r < 8; ++r) {
        float va = sA[r], vp = sP[r];
        if (va > bestA[r]) { bestA[r] = va; idxA[r] = n; }
        if (vp > bestP[r]) { bestP[r] = vp; idxP[r] = n; }
      }
    }
    __syncthreads();          // all waves done reading buf b before refill
  }

  // Reduce across the 16 lanes of each half-wave (masks stay within a half).
  #pragma unroll
  for (int r = 0; r < 8; ++r) {
    #pragma unroll
    for (int off = 8; off >= 1; off >>= 1) {
      float oa = __shfl_xor(bestA[r], off, 32);
      int   ia = __shfl_xor(idxA[r],  off, 32);
      if (oa > bestA[r] || (oa == bestA[r] && ia < idxA[r])) { bestA[r] = oa; idxA[r] = ia; }
      float op = __shfl_xor(bestP[r], off, 32);
      int   ip = __shfl_xor(idxP[r],  off, 32);
      if (op > bestP[r] || (op == bestP[r] && ip < idxP[r])) { bestP[r] = op; idxP[r] = ip; }
    }
  }

  if (l15 == 0) {
    #pragma unroll
    for (int r = 0; r < 8; ++r) {
      outA[row0 + 8 * h + r] = (float)idxA[r];
      outP[row0 + 8 * h + r] = (float)idxP[r];
    }
  }
}

// ---------------------------------------------------------------------------
extern "C" void kernel_launch(void* const* d_in, const int* in_sizes, int n_in,
                              void* d_out, int out_size, void* d_ws, size_t ws_size,
                              hipStream_t stream) {
  const float* x   = (const float*)d_in[0];
  const float* pA  = (const float*)d_in[1];
  const float* pP  = (const float*)d_in[2];
  const float* cbA = (const float*)d_in[3];
  const float* cbP = (const float*)d_in[4];

  char* ws = (char*)d_ws;
  constexpr size_t TAB  = (((size_t)IN_DIM * FREQ * sizeof(float)) + 255) & ~(size_t)255;
  constexpr size_t FEAT = (((size_t)ROWS * VQD * sizeof(unsigned short)) + 255) & ~(size_t)255;
  constexpr size_t CBSZ = (((size_t)NCODE * VQD * sizeof(unsigned short)) + 255) & ~(size_t)255;

  float*          cosT    = (float*)(ws);
  float*          sinT    = (float*)(ws + TAB);
  unsigned short* featA16 = (unsigned short*)(ws + 2 * TAB);
  unsigned short* featP16 = (unsigned short*)(ws + 2 * TAB + FEAT);
  unsigned short* cbA16   = (unsigned short*)(ws + 2 * TAB + 2 * FEAT);
  unsigned short* cbP16   = (unsigned short*)(ws + 2 * TAB + 2 * FEAT + CBSZ);

  build_dft_tables<<<(IN_DIM * FREQ + 255) / 256, 256, 0, stream>>>(cosT, sinT);
  dft_project<<<ROWS, 128, 0, stream>>>(x, pA, pP, cosT, sinT, featA16, featP16);
  normalize_codebook<<<NCODE / 256, 256, 0, stream>>>(cbA, cbA16, NCODE);
  normalize_codebook<<<NCODE / 256, 256, 0, stream>>>(cbP, cbP16, NCODE);

  float* outA = (float*)d_out;          // idx_amp   (19456 floats)
  float* outP = outA + ROWS;            // idx_phase (19456 floats)
  wmma_argmax<<<RTILES / WAVES_PER_BLOCK, WAVES_PER_BLOCK * 32, 0, stream>>>(
      featA16, featP16, cbA16, cbP16, outA, outP);
}
